// SAGENet_51196010169023
// MI455X (gfx1250) — matrix operand order repaired and checked by reading, not compiled, not run on hardware
//
#include <hip/hip_runtime.h>

// ---------------------------------------------------------------------------
// SAGENet (2-layer GraphSAGE, mean aggregation) for MI455X / gfx1250.
// Precision: fp32 gather/scatter + fp32 accumulation; bf16 A/B into
// v_wmma_f32_16x16x32_bf16 for the dense layers (fused bias + activation).
// ---------------------------------------------------------------------------

typedef __attribute__((ext_vector_type(16))) __bf16 v16bf;
typedef __attribute__((ext_vector_type(8)))  __bf16 v8bf;
typedef __attribute__((ext_vector_type(8)))  float  v8f;

#define N_NODES_C 50000
#define IN_C 128
#define K_TOTAL 256   // [mean(128) || self(128)]

// round-to-nearest-even float -> bf16 bits (no __bf16 arithmetic needed)
__device__ __forceinline__ unsigned short f2bf(float f) {
    unsigned int u = __float_as_uint(f);
    u += 0x7FFFu + ((u >> 16) & 1u);
    return (unsigned short)(u >> 16);
}

// ---------------------------------------------------------------------------
__global__ void zero_f32_kernel(float* __restrict__ p, long n) {
    long i = (long)blockIdx.x * blockDim.x + threadIdx.x;
    if (i < n) p[i] = 0.0f;
}

// Pack [W_l ; W_r] (each 128 x outc, row-major) into bf16, transposed to
// wt[col*256 + k] so B-fragments are K-contiguous.
__global__ void wconv_kernel(const float* __restrict__ wl,
                             const float* __restrict__ wr,
                             unsigned short* __restrict__ wt, int outc) {
    int i = blockIdx.x * blockDim.x + threadIdx.x;
    if (i >= outc * K_TOTAL) return;
    int col = i >> 8;          // /256
    int k   = i & 255;
    float v = (k < 128) ? wl[k * outc + col] : wr[(k - 128) * outc + col];
    wt[i] = f2bf(v);           // i == col*256 + k
}

// One wave32 per edge: lane c loads feat[src][4c..4c+3] and atomically adds
// into agg[dst]. Lane 0 optionally bumps the in-degree counter (layer 1 only).
__global__ void scatter_kernel(const float* __restrict__ feat,
                               const int* __restrict__ ei,
                               float* __restrict__ agg,
                               float* __restrict__ cnt,
                               int nEdges, int addCnt) {
    long gid  = (long)blockIdx.x * blockDim.x + threadIdx.x;
    int  edge = (int)(gid >> 5);
    int  lane = (int)(gid & 31);
    if (edge >= nEdges) return;
    int src = ei[edge];
    int dst = ei[nEdges + edge];
    const float4* fr = (const float4*)(feat + (size_t)src * IN_C);
    float4 v = fr[lane];
    float* ad = agg + (size_t)dst * IN_C + lane * 4;
    atomicAdd(ad + 0, v.x);
    atomicAdd(ad + 1, v.y);
    atomicAdd(ad + 2, v.z);
    atomicAdd(ad + 3, v.w);
    if (addCnt && lane == 0) atomicAdd(cnt + dst, 1.0f);
}

// ---------------------------------------------------------------------------
// Fused SAGE layer: out = act( [agg/max(cnt,1) || feat] @ Wt + bias )
// Block = 128 threads (4 waves). 64-row x OUTC-col tile.
// A (64x256) staged in LDS as bf16; B fragments read as 32B vectors from the
// L2-resident bf16 weight pack.
// ---------------------------------------------------------------------------
template <int OUTC, bool SIG>
__global__ __launch_bounds__(128)
void sage_gemm_kernel(const float* __restrict__ feat,
                      const float* __restrict__ agg,
                      const float* __restrict__ cnt,
                      const unsigned short* __restrict__ wt,
                      const float* __restrict__ bias,
                      float* __restrict__ out, int nrows) {
    constexpr int NT = OUTC / 16;
    __shared__ unsigned short As[64 * K_TOTAL];   // 32 KB

    const int tid     = threadIdx.x;
    const int rowBase = blockIdx.x * 64;

    // Stage A block (mean || self) as bf16, coalesced, zero-padded tail.
    for (int i = tid; i < 64 * K_TOTAL; i += 128) {
        int r  = i >> 8;
        int k  = i & 255;
        int gr = rowBase + r;
        float v = 0.0f;
        if (gr < nrows) {
            if (k < 128)
                v = agg[(size_t)gr * IN_C + k] / fmaxf(cnt[gr], 1.0f);
            else
                v = feat[(size_t)gr * IN_C + (k - 128)];
        }
        As[i] = f2bf(v);
    }
    __syncthreads();

    const int lane  = tid & 31;
    const int wv    = tid >> 5;            // wave id: 16-row strip
    const int laneM = lane & 15;
    const int kSelA = (lane >> 4) << 4;    // A: +0 or +16 (K halves across lanes)
    const int kSelB = (lane >> 4) << 4;    // B: +0 or +16 (K halves across lanes)

    const unsigned short* AsRow = &As[(wv * 16 + laneM) * K_TOTAL];

    v8f zero = {0.f, 0.f, 0.f, 0.f, 0.f, 0.f, 0.f, 0.f};
    v8f acc[NT];
#pragma unroll
    for (int nt = 0; nt < NT; ++nt) acc[nt] = zero;

#pragma unroll
    for (int kk = 0; kk < K_TOTAL; kk += 32) {
        // A fragment: elems 0..7 -> K = kk+kSelA+e ; 8..15 -> K = kk+8+kSelA+e
        v8bf alo = *(const v8bf*)(AsRow + kk + kSelA);
        v8bf ahi = *(const v8bf*)(AsRow + kk + 8 + kSelA);
        v16bf a = __builtin_shufflevector(alo, ahi,
                    0, 1, 2, 3, 4, 5, 6, 7, 8, 9, 10, 11, 12, 13, 14, 15);
#pragma unroll
        for (int nt = 0; nt < NT; ++nt) {
            int col = nt * 16 + laneM;
            v16bf b = *(const v16bf*)(wt + (size_t)col * K_TOTAL + kk + kSelB);
            acc[nt] = __builtin_amdgcn_wmma_f32_16x16x32_bf16(
                false, a, false, b, (short)0, acc[nt], false, false);
        }
    }

    // Epilogue: C layout — lane<16: N=lane, M=v ; lane>=16: N=lane-16, M=8+v
    const int rOff = (lane >> 4) << 3;
#pragma unroll
    for (int nt = 0; nt < NT; ++nt) {
#pragma unroll
        for (int v = 0; v < 8; ++v) {
            int r = rowBase + wv * 16 + rOff + v;
            int c = nt * 16 + laneM;
            if (r < nrows) {
                float val = acc[nt][v] + bias[c];
                val = SIG ? 1.0f / (1.0f + __expf(-val)) : fmaxf(val, 0.0f);
                out[(size_t)r * OUTC + c] = val;
            }
        }
    }
}

// ---------------------------------------------------------------------------
extern "C" void kernel_launch(void* const* d_in, const int* in_sizes, int n_in,
                              void* d_out, int out_size, void* d_ws, size_t ws_size,
                              hipStream_t stream) {
    const float* x   = (const float*)d_in[0];
    const int*   ei  = (const int*)  d_in[1];   // [2, E] int32: row0=src, row1=dst
    const float* W1l = (const float*)d_in[2];
    const float* b1  = (const float*)d_in[3];
    const float* W1r = (const float*)d_in[4];
    const float* W2l = (const float*)d_in[5];
    const float* b2  = (const float*)d_in[6];
    const float* W2r = (const float*)d_in[7];
    float* out = (float*)d_out;

    const int N = in_sizes[0] / IN_C;           // 50000
    const int E = in_sizes[1] / 2;              // 800000

    // workspace layout (fp32 unless noted)
    float* agg  = (float*)d_ws;                           // N*128
    float* hbuf = agg + (size_t)N * IN_C;                 // N*128
    float* cnt  = hbuf + (size_t)N * IN_C;                // N
    unsigned short* w1 = (unsigned short*)(cnt + N);      // 256*128 bf16
    unsigned short* w2 = w1 + K_TOTAL * 128;              // 256*64  bf16

    const int threads = 256;

    // 1) weight packs (bf16, transposed)
    wconv_kernel<<<(K_TOTAL * 128 + threads - 1) / threads, threads, 0, stream>>>(
        W1l, W1r, w1, 128);
    wconv_kernel<<<(K_TOTAL * 64 + threads - 1) / threads, threads, 0, stream>>>(
        W2l, W2r, w2, 64);

    // 2) zero agg + cnt
    long aggN = (long)N * IN_C;
    zero_f32_kernel<<<(unsigned)((aggN + threads - 1) / threads), threads, 0, stream>>>(agg, aggN);
    zero_f32_kernel<<<(N + threads - 1) / threads, threads, 0, stream>>>(cnt, N);

    // 3) layer-1 scatter (also builds in-degree counts)
    long sThreads = (long)E * 32;
    unsigned sBlocks = (unsigned)((sThreads + threads - 1) / threads);
    scatter_kernel<<<sBlocks, threads, 0, stream>>>(x, ei, agg, cnt, E, 1);

    // 4) layer 1 GEMM + ReLU -> h
    unsigned gBlocks = (unsigned)((N + 63) / 64);
    sage_gemm_kernel<128, false><<<gBlocks, 128, 0, stream>>>(
        x, agg, cnt, w1, b1, hbuf, N);

    // 5) re-zero agg, layer-2 scatter over h (reuse cnt: same graph)
    zero_f32_kernel<<<(unsigned)((aggN + threads - 1) / threads), threads, 0, stream>>>(agg, aggN);
    scatter_kernel<<<sBlocks, threads, 0, stream>>>(hbuf, ei, agg, cnt, E, 0);

    // 6) layer 2 GEMM + sigmoid -> out
    sage_gemm_kernel<64, true><<<gBlocks, 128, 0, stream>>>(
        hbuf, agg, cnt, w2, b2, out, N);
}
